// MultiHeadAttention_3341484556859
// MI455X (gfx1250) — compile-verified, matrix operand order checked by reading, not run on hardware
//
#include <hip/hip_runtime.h>
#include <hip/hip_bf16.h>
#include <math.h>

// Problem constants (from reference): B=2, S=2048, E=1024, H=16, D=64
constexpr int CB = 2;
constexpr int CS = 2048;
constexpr int CE = 1024;
constexpr int CH = 16;
constexpr int CD = 64;
constexpr int CHD = CH * CD; // 1024

typedef _Float16 half_t;
typedef __attribute__((ext_vector_type(16))) _Float16 v16h;
typedef __attribute__((ext_vector_type(8)))  _Float16 v8h;
typedef __attribute__((ext_vector_type(8)))  float    v8f;

// ---------------------------------------------------------------------------
// WMMA helper: D = A(16x32 f16) * B(32x16 f16) + C(16x16 f32)
// ---------------------------------------------------------------------------
__device__ __forceinline__ v8f wmma_f16(v16h a, v16h b, v8f c) {
  return __builtin_amdgcn_wmma_f32_16x16x32_f16(
      /*neg_a=*/false, a, /*neg_b=*/false, b,
      /*c_mod=*/(short)0, c, /*reuse_a=*/false, /*reuse_b=*/false);
}

// ---------------------------------------------------------------------------
// Unified fragment loader.
// Fragment element (sel, k) = p[sel*ld + k], sel = lane&15, k in [0,32).
// Lane (sel, h=lane>>4) holds halves k = h*8..h*8+7 and 16+h*8..+7
// -> two contiguous 16-byte loads. Requires 16B alignment, ld % 8 == 0.
// ---------------------------------------------------------------------------
__device__ __forceinline__ v16h load_frag(const half_t* __restrict__ p, int ld) {
  const int lane = threadIdx.x & 31;
  const int sel  = lane & 15;
  const int hh   = lane >> 4;
  const v8h* q = (const v8h*)(p + (size_t)sel * ld + hh * 8);
  union { v16h v; v8h h[2]; } u;
  u.h[0] = q[0];
  u.h[1] = q[2];
  return u.v;
}

// ---------------------------------------------------------------------------
// Async global -> LDS copy, 16 bytes per lane (ASYNCcnt-tracked).
// ---------------------------------------------------------------------------
__device__ __forceinline__ void async_copy_b128(const half_t* lds_dst,
                                                const half_t* gsrc) {
  unsigned int loff = (unsigned int)(size_t)(const void*)lds_dst;
  asm volatile("global_load_async_to_lds_b128 %0, %1, off"
               :
               : "v"(loff), "v"(gsrc)
               : "memory");
}

__device__ __forceinline__ void wait_async0() {
  asm volatile("s_wait_asynccnt 0x0" ::: "memory");
}

// ---------------------------------------------------------------------------
// Converts
// ---------------------------------------------------------------------------
__global__ void f32_to_f16_kernel(const float* __restrict__ in,
                                  half_t* __restrict__ out, int n) {
  int i = blockIdx.x * blockDim.x + threadIdx.x;
  if (i < n) out[i] = (half_t)in[i];
}

// Wq/Wk/Wv [H,E,D] f32 -> [H,D,E] f16 (K-contiguous per output column)
__global__ void cvt_tr_wqkv_kernel(const float* __restrict__ in,
                                   half_t* __restrict__ out) {
  int i = blockIdx.x * 256 + threadIdx.x;   // over H*E*D = 1Mi, exact grid
  int hd = i >> 16;            // / (E*D)
  int r  = i & 65535;
  int e  = r >> 6;             // / D
  int d  = r & 63;
  out[((size_t)hd * CD + d) * CE + e] = (half_t)in[i];
}

// Wo [H*D, E] f32 -> [E, H*D] f16
__global__ void cvt_tr_wo_kernel(const float* __restrict__ in,
                                 half_t* __restrict__ out) {
  int i = blockIdx.x * 256 + threadIdx.x;   // over HD*E = 1Mi, exact grid
  int k = i >> 10;             // / E
  int e = i & 1023;
  out[(size_t)e * CHD + k] = (half_t)in[i];
}

// ---------------------------------------------------------------------------
// QKV projection: out = x . W + bias, one head per wave, 16x64 strip.
// grid = (B*S/16, H/4); block = 128 (4 waves).
// TR_OUT=false: out [B,H,S,D] (Q, K); TR_OUT=true: out [B,H,D,S] (V).
// ---------------------------------------------------------------------------
template <bool TR_OUT>
__global__ void __launch_bounds__(128)
qkv_proj_kernel(const half_t* __restrict__ xh,   // [B*S, E] f16
                const half_t* __restrict__ WT,   // [H, D, E] f16 (transposed)
                const float*  __restrict__ bias, // [H, D] f32
                half_t* __restrict__ outp)
{
  const int wid  = threadIdx.x >> 5;
  const int hd   = blockIdx.y * 4 + wid;   // head
  const int row0 = blockIdx.x * 16;        // global row in [0, B*S)
  const int bb   = row0 / CS;
  const int s0   = row0 % CS;

  const half_t* abase = xh + (size_t)row0 * CE;
  const half_t* wbase = WT + (size_t)hd * CD * CE;

  v8f acc[4] = {{}, {}, {}, {}};
  for (int k = 0; k < CE; k += 32) {
    __builtin_prefetch((const void*)(abase + k + 256), 0, 1);
    const v16h af = load_frag(abase + k, CE);
#pragma unroll
    for (int j = 0; j < 4; ++j) {
      const v16h bf = load_frag(wbase + (size_t)(j * 16) * CE + k, CE);
      acc[j] = wmma_f16(af, bf, acc[j]);
    }
  }

  const int lane = threadIdx.x & 31;
  const int n = lane & 15;
  const int h = lane >> 4;
  if (TR_OUT) {
#pragma unroll
    for (int j = 0; j < 4; ++j) {
      const float bv = bias[hd * CD + j * 16 + n];
      v8h pk;
#pragma unroll
      for (int r = 0; r < 8; ++r) pk[r] = (half_t)(acc[j][r] + bv);
      half_t* ob = outp + ((size_t)(bb * CH + hd) * CD + j * 16 + n) * CS
                        + s0 + 8 * h;
      *(v8h*)ob = pk; // 8 consecutive s-positions -> one b128 store
    }
  } else {
#pragma unroll
    for (int j = 0; j < 4; ++j) {
      const float bv = bias[hd * CD + j * 16 + n];
      half_t* ob = outp + ((size_t)(bb * CH + hd) * CS + s0) * CD + j * 16 + n;
#pragma unroll
      for (int r = 0; r < 8; ++r)
        ob[(size_t)(r + 8 * h) * CD] = (half_t)(acc[j][r] + bv);
    }
  }
}

// ---------------------------------------------------------------------------
// Fused causal flash attention with async double-buffered K/V staging.
// grid = (S/64, B*H); block = 128 (4 waves x 16 q-rows each).
// Q,K row-major [B,H,S,D]; V transposed [B,H,D,S].
// Per 32-key step: whole block stages K(32x64) + V^T(64x32) tiles into LDS
// via GLOBAL_LOAD_ASYNC_TO_LDS_B128 while WMMAs consume the previous tile.
// ---------------------------------------------------------------------------
__global__ void __launch_bounds__(128)
flash_attn_kernel(const half_t* __restrict__ qh,  // [B,H,S,D]
                  const half_t* __restrict__ kh,  // [B,H,S,D]
                  const half_t* __restrict__ vt,  // [B,H,D,S]
                  half_t* __restrict__ oh)        // [B*S, H*D]
{
  __shared__ alignas(16) half_t Kbuf[2][32 * 64]; // [row s][d], ld=64
  __shared__ alignas(16) half_t Vbuf[2][64 * 32]; // [d][row s], ld=32
  __shared__ alignas(16) half_t plds[4][16 * 32]; // per-wave P scratch

  const int wid  = threadIdx.x >> 5;
  const int lane = threadIdx.x & 31;
  const int n    = lane & 15;
  const int h    = lane >> 4;
  const int tid  = threadIdx.x;

  const int bh = blockIdx.y;                 // b*H + hd
  const int bb = bh / CH;
  const int hd = bh % CH;
  const int q0 = blockIdx.x * 64 + wid * 16; // this wave's first q row

  const half_t* qbase = qh + ((size_t)bh * CS + q0) * CD;
  const half_t* kbase = kh + (size_t)bh * CS * CD;
  const half_t* vbase = vt + (size_t)bh * CD * CS;

  // Q fragments for the two 32-wide halves of D=64 (loaded once).
  const v16h qa0 = load_frag(qbase, CD);
  const v16h qa1 = load_frag(qbase + 32, CD);

  v8f o0 = {}, o1 = {}, o2 = {}, o3 = {};
  float mrow[8], lrow[8];
#pragma unroll
  for (int r = 0; r < 8; ++r) { mrow[r] = -1e30f; lrow[r] = 0.0f; }

  half_t* pw = &plds[wid][0];
  const float scale = 0.125f; // 1/sqrt(64)

  // Stage one 32-key step (8 KB) into LDS buffer `buf`: all 128 threads,
  // 4 async b128 instructions each.
  auto stage = [&](int c0, int buf) {
    // K tile: rows c0..c0+31 of [S,D] -> 4 KB contiguous
    const half_t* kg = kbase + (size_t)c0 * CD;
    half_t* kl = &Kbuf[buf][0];
#pragma unroll
    for (int i = 0; i < 2; ++i) {
      int c = tid + i * 128;              // 16B chunk id, 0..255
      async_copy_b128(kl + c * 8, kg + c * 8);
    }
    // V^T tile: 64 rows of 32 halves (64B), row stride CS
    half_t* vl = &Vbuf[buf][0];
#pragma unroll
    for (int i = 0; i < 2; ++i) {
      int c = tid + i * 128;              // 0..255
      int row = c >> 2;                   // d
      int off = (c & 3) * 8;              // halves within row
      async_copy_b128(vl + row * 32 + off,
                      vbase + (size_t)row * CS + c0 + off);
    }
  };

  // Trip counts: block-uniform outer bound, wave-uniform causal bound.
  const int steps_max = 2 * blockIdx.x + 2;      // covers cols [0, q0_blk+64)
  const int steps_w   = (q0 + 16 + 31) / 32;     // this wave needs [0, q0+16)

  stage(0, 0); // prologue

  for (int st = 0; st < steps_max; ++st) {
    wait_async0();       // own async loads complete
    __syncthreads();     // everyone's tile data visible
    if (st + 1 < steps_max)
      stage((st + 1) * 32, (st + 1) & 1);

    if (st < steps_w) {
      const int c0 = st * 32;
      const int buf = st & 1;
      const half_t* kl = &Kbuf[buf][0];
      const half_t* vl = &Vbuf[buf][0];

      // ---- scores S[16 x 32] = Q * K^T (4 WMMAs from LDS) ----
      const v16h kb0a = load_frag(kl, CD);
      const v16h kb0b = load_frag(kl + 32, CD);
      const v16h kb1a = load_frag(kl + 16 * CD, CD);
      const v16h kb1b = load_frag(kl + 16 * CD + 32, CD);
      v8f sA = {}, sB = {};
      sA = wmma_f16(qa0, kb0a, sA);
      sA = wmma_f16(qa1, kb0b, sA);
      sB = wmma_f16(qa0, kb1a, sB);
      sB = wmma_f16(qa1, kb1b, sB);

      // ---- causal mask, online softmax, stash P in LDS ----
#pragma unroll
      for (int r = 0; r < 8; ++r) {
        const int qrow = q0 + r + 8 * h;
        float a0 = sA[r] * scale; if (c0 + n      > qrow) a0 = -1e30f;
        float a1 = sB[r] * scale; if (c0 + 16 + n > qrow) a1 = -1e30f;

        float mx = fmaxf(a0, a1);
#pragma unroll
        for (int off = 1; off < 16; off <<= 1)
          mx = fmaxf(mx, __shfl_xor(mx, off, 16));
        const float nm = fmaxf(mrow[r], mx);

        const float p0 = __expf(a0 - nm);
        const float p1 = __expf(a1 - nm);
        float rs = p0 + p1;
#pragma unroll
        for (int off = 1; off < 16; off <<= 1)
          rs += __shfl_xor(rs, off, 16);

        const float corr = __expf(mrow[r] - nm);
        lrow[r] = lrow[r] * corr + rs;
        mrow[r] = nm;
        o0[r] *= corr; o1[r] *= corr; o2[r] *= corr; o3[r] *= corr;

        const int m = r + 8 * h;
        pw[m * 32 + n]      = (half_t)p0;
        pw[m * 32 + 16 + n] = (half_t)p1;
      }

      // Wave-local LDS RAW fence (cross-lane through LDS).
      __builtin_amdgcn_wave_barrier();
      asm volatile("s_wait_dscnt 0x0" ::: "memory");
      __builtin_amdgcn_wave_barrier();

      // ---- O += P(16x32) * V(32x64) (4 WMMAs from LDS) ----
      const v16h pa = load_frag(pw, 32);
      o0 = wmma_f16(pa, load_frag(vl +  0 * 32, 32), o0);
      o1 = wmma_f16(pa, load_frag(vl + 16 * 32, 32), o1);
      o2 = wmma_f16(pa, load_frag(vl + 32 * 32, 32), o2);
      o3 = wmma_f16(pa, load_frag(vl + 48 * 32, 32), o3);
    }
  }

  // ---- normalize and store concat-head f16 output ----
#pragma unroll
  for (int r = 0; r < 8; ++r) {
    const float inv = 1.0f / lrow[r];
    const int m = r + 8 * h;
    half_t* ob = oh + ((size_t)(bb * CS) + q0 + m) * CHD + hd * CD + n;
    ob[0]  = (half_t)(o0[r] * inv);
    ob[16] = (half_t)(o1[r] * inv);
    ob[32] = (half_t)(o2[r] * inv);
    ob[48] = (half_t)(o3[r] * inv);
  }
}

// ---------------------------------------------------------------------------
// Output projection: out[B*S, E] f32 = A[B*S, H*D] f16 . Wo + bo
// WoT is [E, H*D] (K-contiguous). One 16x64 strip per wave.
// grid = (B*S/16, 4); block = 128.
// ---------------------------------------------------------------------------
__global__ void __launch_bounds__(128)
out_proj_kernel(const half_t* __restrict__ ah,   // [B*S, H*D]
                const half_t* __restrict__ WoT,  // [E, H*D]
                const float*  __restrict__ bo,   // [E]
                float* __restrict__ out)         // [B*S, E]
{
  const int wid  = threadIdx.x >> 5;
  const int si   = blockIdx.y * 4 + wid;  // strip 0..15
  const int row0 = blockIdx.x * 16;
  const int col0 = si * 64;

  const half_t* abase = ah + (size_t)row0 * CHD;

  v8f acc[4] = {{}, {}, {}, {}};
  for (int k = 0; k < CHD; k += 32) {
    __builtin_prefetch((const void*)(abase + k + 256), 0, 1);
    const v16h af = load_frag(abase + k, CHD);
#pragma unroll
    for (int j = 0; j < 4; ++j) {
      const v16h bf = load_frag(WoT + (size_t)(col0 + 16 * j) * CHD + k, CHD);
      acc[j] = wmma_f16(af, bf, acc[j]);
    }
  }

  const int lane = threadIdx.x & 31;
  const int n = lane & 15;
  const int h = lane >> 4;
#pragma unroll
  for (int j = 0; j < 4; ++j) {
    const float bv = bo[col0 + 16 * j + n];
    float* ob = out + (size_t)row0 * CE + col0 + 16 * j + n;
#pragma unroll
    for (int r = 0; r < 8; ++r)
      ob[(size_t)(r + 8 * h) * CE] = acc[j][r] + bv;
  }
}

// ---------------------------------------------------------------------------
// Host launch
// ---------------------------------------------------------------------------
extern "C" void kernel_launch(void* const* d_in, const int* in_sizes, int n_in,
                              void* d_out, int out_size, void* d_ws, size_t ws_size,
                              hipStream_t stream) {
  const float* x  = (const float*)d_in[0];
  const float* Wq = (const float*)d_in[1];
  const float* Wk = (const float*)d_in[2];
  const float* Wv = (const float*)d_in[3];
  const float* bq = (const float*)d_in[4];
  const float* bk = (const float*)d_in[5];
  const float* bv = (const float*)d_in[6];
  const float* Wo = (const float*)d_in[7];
  const float* bo = (const float*)d_in[8];
  float* out = (float*)d_out;

  // Workspace (all f16): xh | WqT | WkT | WvT | WoT | qh | kh | vT | ah
  half_t* w = (half_t*)d_ws;
  size_t off = 0;
  const size_t nX = (size_t)CB * CS * CE;      // 4 Mi
  const size_t nW = (size_t)CH * CE * CD;      // 1 Mi
  const size_t nQ = (size_t)CB * CH * CS * CD; // 4 Mi
  half_t* xh  = w + off; off += nX;
  half_t* WqT = w + off; off += nW;
  half_t* WkT = w + off; off += nW;
  half_t* WvT = w + off; off += nW;
  half_t* WoT = w + off; off += nW;
  half_t* qh  = w + off; off += nQ;
  half_t* kh  = w + off; off += nQ;
  half_t* vT  = w + off; off += nQ;
  half_t* ah  = w + off; off += nX;            // ~48 MiB total

  // 1) convert (+transpose weights) to f16
  f32_to_f16_kernel<<<(int)(nX / 256), 256, 0, stream>>>(x, xh, (int)nX);
  cvt_tr_wqkv_kernel<<<(int)(nW / 256), 256, 0, stream>>>(Wq, WqT);
  cvt_tr_wqkv_kernel<<<(int)(nW / 256), 256, 0, stream>>>(Wk, WkT);
  cvt_tr_wqkv_kernel<<<(int)(nW / 256), 256, 0, stream>>>(Wv, WvT);
  cvt_tr_wo_kernel  <<<(int)(nW / 256), 256, 0, stream>>>(Wo, WoT);

  // 2) QKV projections (WMMA GEMMs); V written transposed [B,H,D,S]
  dim3 gp(CB * CS / 16, CH / 4);
  qkv_proj_kernel<false><<<gp, 128, 0, stream>>>(xh, WqT, bq, qh);
  qkv_proj_kernel<false><<<gp, 128, 0, stream>>>(xh, WkT, bk, kh);
  qkv_proj_kernel<true ><<<gp, 128, 0, stream>>>(xh, WvT, bv, vT);

  // 3) fused causal flash attention (async-staged K/V)
  flash_attn_kernel<<<dim3(CS / 64, CB * CH), 128, 0, stream>>>(qh, kh, vT, ah);

  // 4) output projection
  out_proj_kernel<<<dim3(CB * CS / 16, 4), 128, 0, stream>>>(ah, WoT, bo, out);
}
